// TransCN_6597069766689
// MI455X (gfx1250) — matrix operand order repaired and checked by reading, not compiled, not run on hardware
//
#include <hip/hip_runtime.h>

#define NN 50000   // nodes
#define EE 400000  // edges
#define DD 128     // feature dim
#define NG 256     // graphs
#define NC 20      // classes

typedef __attribute__((ext_vector_type(16))) _Float16 v16h;
typedef __attribute__((ext_vector_type(8)))  float    v8f;

// ---- ordered-int encoding so atomicMax(int) == atomicMax(float) ----
__device__ __forceinline__ int f32_to_ordered(float f) {
    int i = __float_as_int(f);
    return i >= 0 ? i : (i ^ 0x7fffffff);
}
__device__ __forceinline__ float ordered_to_f32(int k) {
    return __int_as_float(k >= 0 ? k : (k ^ 0x7fffffff));
}

// ---------------------------------------------------------------------------
// Pack 4 fp32 [128x128] weights (row-major d_in x d_out) into f16 WMMA
// B-fragment layout: frag[mat][kt][nt][lane][j], where, in the contiguous k order used here,
// B element j of lane l holds K = 32*kt + 16*(l>>4) + j, N = 16*nt + (l&15).
// Each lane then loads one contiguous v16h (32B) per fragment in the GEMM.
// ---------------------------------------------------------------------------
__global__ void k_pack_w(const float* __restrict__ Wq, const float* __restrict__ Wk,
                         const float* __restrict__ Wv, const float* __restrict__ Ws,
                         _Float16* __restrict__ wf) {
    int tid  = blockIdx.x * blockDim.x + threadIdx.x;   // 65536 total
    int j    = tid & 15;
    int lane = (tid >> 4) & 31;
    int nt   = (tid >> 9) & 7;
    int kt   = (tid >> 12) & 3;
    int mat  = tid >> 14;
    const float* W = (mat == 0) ? Wq : (mat == 1) ? Wk : (mat == 2) ? Wv : Ws;
    int K = kt * 32 + (lane >> 4) * 16 + j;
    int N = nt * 16 + (lane & 15);
    wf[tid] = (_Float16)W[K * DD + N];
}

// ---------------------------------------------------------------------------
// Fused Q/K/V/S GEMM: [NN x 128] @ [128 x 128] + bias, blockIdx.y selects matrix.
// 128 threads = 4 waves; wave w computes rows [16*bx,16*bx+16) x cols [32w,32w+32).
// Activation tile staged in LDS as f16; A fragment in contiguous k order (same as packed B):
//   lane l: M = l&15, h = l>>4;  j in 0..15 -> K = 32*kt + 16h + j
// i.e. two contiguous 16B LDS chunks per k-step.
// ---------------------------------------------------------------------------
__global__ __launch_bounds__(128) void k_gemm(
        const float* __restrict__ X, const _Float16* __restrict__ wf,
        const float* __restrict__ bq, const float* __restrict__ bk,
        const float* __restrict__ bv, const float* __restrict__ bs,
        float* __restrict__ Q, float* __restrict__ Kout,
        float* __restrict__ V, float* __restrict__ OUT) {
    __shared__ _Float16 lds[16 * 136];   // 16 rows x 128 halves, +8 half pad/row
    int t   = threadIdx.x;
    int bx  = blockIdx.x;
    int mat = blockIdx.y;

    // stage 16x128 activation tile, fp32 -> f16
    {
        int i  = t >> 3;           // row 0..15
        int c0 = (t & 7) * 16;     // 16 consecutive cols
        const float4* srcp = (const float4*)(X + (size_t)(bx * 16 + i) * DD + c0);
        _Float16* dstp = lds + i * 136 + c0;
#pragma unroll
        for (int u = 0; u < 4; ++u) {
            float4 f = srcp[u];
            dstp[u * 4 + 0] = (_Float16)f.x;
            dstp[u * 4 + 1] = (_Float16)f.y;
            dstp[u * 4 + 2] = (_Float16)f.z;
            dstp[u * 4 + 3] = (_Float16)f.w;
        }
    }
    __syncthreads();

    int lane = t & 31, w = t >> 5;
    int m = lane & 15, h = lane >> 4;
    const v16h* wfv = (const v16h*)wf;

    v8f acc0 = {}, acc1 = {};
#pragma unroll
    for (int kt = 0; kt < 4; ++kt) {
        v16h a;
        const _Float16* p0 = lds + m * 136 + kt * 32 + h * 16;
        const _Float16* p1 = p0 + 8;
#pragma unroll
        for (int j = 0; j < 8; ++j) { a[j] = p0[j]; a[j + 8] = p1[j]; }
        v16h b0 = wfv[(size_t)(((mat * 4 + kt) * 8 + 2 * w) * 32 + lane)];
        v16h b1 = wfv[(size_t)(((mat * 4 + kt) * 8 + 2 * w + 1) * 32 + lane)];
        acc0 = __builtin_amdgcn_wmma_f32_16x16x32_f16(false, a, false, b0,
                                                      (short)0, acc0, false, false);
        acc1 = __builtin_amdgcn_wmma_f32_16x16x32_f16(false, a, false, b1,
                                                      (short)0, acc1, false, false);
    }

    const float* bias = (mat == 0) ? bq : (mat == 1) ? bk : (mat == 2) ? bv : bs;
    float* O = (mat == 0) ? Q : (mat == 1) ? Kout : (mat == 2) ? V : OUT;
    int n0 = 32 * w + (lane & 15);
    int n1 = n0 + 16;
    float bb0 = bias[n0], bb1 = bias[n1];
    // D layout: reg r, lanes 0-15 -> M=r, lanes 16-31 -> M=r+8; N = lane&15
#pragma unroll
    for (int r = 0; r < 8; ++r) {
        size_t row = (size_t)(bx * 16 + r + 8 * h);
        O[row * DD + n0] = acc0[r] + bb0;
        O[row * DD + n1] = acc1[r] + bb1;
    }
}

// ---------------------------------------------------------------------------
// Edge logits: one wave per edge, float4 per lane, shuffle reduce,
// then segment-max via ordered-int atomicMax.
// ---------------------------------------------------------------------------
__global__ void k_logits(const float* __restrict__ Q, const float* __restrict__ Km,
                         const int* __restrict__ src, const int* __restrict__ dst,
                         float* __restrict__ logits, int* __restrict__ maxb) {
    int e    = blockIdx.x * 8 + (threadIdx.x >> 5);
    int lane = threadIdx.x & 31;
    int s = src[e], d = dst[e];
    float4 a = ((const float4*)(Q  + (size_t)d * DD))[lane];
    float4 b = ((const float4*)(Km + (size_t)s * DD))[lane];
    float v = a.x * b.x + a.y * b.y + a.z * b.z + a.w * b.w;
    for (int o = 16; o > 0; o >>= 1) v += __shfl_xor(v, o, 32);
    if (lane == 0) {
        v *= 0.08838834764831845f;   // 1/sqrt(128)
        logits[e] = v;
        atomicMax(maxb + d, f32_to_ordered(v));
    }
}

__global__ void k_exp(const float* __restrict__ logits, const int* __restrict__ maxb,
                      const int* __restrict__ dst, float* __restrict__ wexp,
                      float* __restrict__ denom) {
    int e = blockIdx.x * blockDim.x + threadIdx.x;
    if (e >= EE) return;
    int d = dst[e];
    float m  = ordered_to_f32(maxb[d]);
    float ex = __expf(logits[e] - m);
    wexp[e] = ex;
    atomicAdd(denom + d, ex);
}

__global__ void k_scatter(const float* __restrict__ wexp, const float* __restrict__ denom,
                          const float* __restrict__ V, const int* __restrict__ src,
                          const int* __restrict__ dst, float* __restrict__ OUT) {
    int e    = blockIdx.x * 8 + (threadIdx.x >> 5);
    int lane = threadIdx.x & 31;
    int s = src[e], d = dst[e];
    float alpha = wexp[e] / (denom[d] + 1e-16f);
    float4 vv = ((const float4*)(V + (size_t)s * DD))[lane];
    float* o = OUT + (size_t)d * DD + lane * 4;
    atomicAdd(o + 0, alpha * vv.x);
    atomicAdd(o + 1, alpha * vv.y);
    atomicAdd(o + 2, alpha * vv.z);
    atomicAdd(o + 3, alpha * vv.w);
}

// ---------------------------------------------------------------------------
// BatchNorm: per-column sum / sumsq.  Each thread owns one column (t&127),
// privately accumulates 32 rows, folds into LDS via ds_add_f32, then one
// global atomic per block per column.
// ---------------------------------------------------------------------------
__global__ void k_bn_stats(const float* __restrict__ X, float* __restrict__ bnsum) {
    __shared__ float ls[256];
    int t = threadIdx.x;
    int c = t & 127;
    ls[t] = 0.0f;
    __syncthreads();
    float s1 = 0.0f, s2 = 0.0f;
    int r0 = blockIdx.x * 64;
#pragma unroll 4
    for (int k = 0; k < 32; ++k) {
        int row = r0 + (t >> 7) + 2 * k;
        if (row < NN) {
            float v = X[(size_t)row * DD + c];
            s1 += v;
            s2 += v * v;
        }
    }
    atomicAdd(&ls[c], s1);
    atomicAdd(&ls[128 + c], s2);
    __syncthreads();
    if (t < 128) {
        atomicAdd(bnsum + t, ls[t]);
        atomicAdd(bnsum + 128 + t, ls[128 + t]);
    }
}

__global__ void k_bn_coef(const float* __restrict__ bnsum, const float* __restrict__ gamma,
                          const float* __restrict__ beta, float* __restrict__ coef) {
    int c = threadIdx.x;   // 128 threads
    float invn = 1.0f / (float)NN;
    float mean = bnsum[c] * invn;
    float var  = bnsum[128 + c] * invn - mean * mean;
    float sc   = gamma[c] * __frsqrt_rn(var + 1e-5f);
    coef[c]       = sc;
    coef[128 + c] = beta[c] - mean * sc;
}

__global__ void k_bn_relu(const float* __restrict__ X, const float* __restrict__ coef,
                          float* __restrict__ H) {
    int idx = blockIdx.x * blockDim.x + threadIdx.x;
    int c = idx & 127;
    float v = X[idx] * coef[c] + coef[128 + c];
    H[idx] = v > 0.0f ? v : 0.0f;
}

// ---------------------------------------------------------------------------
// Global mean pool + linear head
// ---------------------------------------------------------------------------
__global__ void k_pool(const float* __restrict__ H, const int* __restrict__ batch,
                       float* __restrict__ gsum, float* __restrict__ gcnt) {
    int idx = blockIdx.x * blockDim.x + threadIdx.x;  // NN*DD threads
    int i = idx >> 7, c = idx & 127;
    int b = batch[i];
    atomicAdd(gsum + (size_t)b * DD + c, H[idx]);
    if (c == 0) atomicAdd(gcnt + b, 1.0f);
}

__global__ void k_head(const float* __restrict__ gsum, const float* __restrict__ gcnt,
                       const float* __restrict__ Wl, const float* __restrict__ bl,
                       float* __restrict__ out) {
    int g = blockIdx.x, lane = threadIdx.x;   // 32 threads
    float4 sv = ((const float4*)(gsum + (size_t)g * DD))[lane];
    float cnt = gcnt[g];
    cnt = cnt > 1.0f ? cnt : 1.0f;
    float inv = 1.0f / cnt;
    float m0 = sv.x * inv, m1 = sv.y * inv, m2 = sv.z * inv, m3 = sv.w * inv;
    int k0 = lane * 4;
    for (int c = 0; c < NC; ++c) {
        float p = m0 * Wl[(k0 + 0) * NC + c] + m1 * Wl[(k0 + 1) * NC + c] +
                  m2 * Wl[(k0 + 2) * NC + c] + m3 * Wl[(k0 + 3) * NC + c];
        for (int o = 16; o > 0; o >>= 1) p += __shfl_xor(p, o, 32);
        if (lane == 0) out[g * NC + c] = p + bl[c];
    }
}

// ---------------------------------------------------------------------------
// Utility init kernels
// ---------------------------------------------------------------------------
__global__ void k_zero(float* __restrict__ p, int n) {
    int i = blockIdx.x * blockDim.x + threadIdx.x;
    if (i < n) p[i] = 0.0f;
}
__global__ void k_fill_int(int* __restrict__ p, int v, int n) {
    int i = blockIdx.x * blockDim.x + threadIdx.x;
    if (i < n) p[i] = v;
}

// ---------------------------------------------------------------------------
extern "C" void kernel_launch(void* const* d_in, const int* in_sizes, int n_in,
                              void* d_out, int out_size, void* d_ws, size_t ws_size,
                              hipStream_t stream) {
    const float* x     = (const float*)d_in[0];
    const int*   ei    = (const int*)d_in[1];
    const int*   src   = ei;            // edge_index[0]
    const int*   dst   = ei + EE;       // edge_index[1]
    const int*   batch = (const int*)d_in[2];
    const float* Wlin  = (const float*)d_in[33];
    const float* blin  = (const float*)d_in[34];

    // workspace carve-out (256B aligned slots)
    char*  base = (char*)d_ws;
    size_t o = 0;
    auto alloc = [&](size_t bytes) -> void* {
        void* p = base + o;
        o += (bytes + 255) & ~(size_t)255;
        return p;
    };
    float*    Q      = (float*)alloc((size_t)NN * DD * 4);
    float*    Km     = (float*)alloc((size_t)NN * DD * 4);
    float*    V      = (float*)alloc((size_t)NN * DD * 4);
    float*    OUT    = (float*)alloc((size_t)NN * DD * 4);  // root term + aggregation
    float*    H      = (float*)alloc((size_t)NN * DD * 4);  // post BN+ReLU
    float*    logits = (float*)alloc((size_t)EE * 4);
    float*    wexp   = (float*)alloc((size_t)EE * 4);
    int*      maxb   = (int*)alloc((size_t)NN * 4);
    float*    denom  = (float*)alloc((size_t)NN * 4);
    float*    bnsum  = (float*)alloc(256 * 4);
    float*    coef   = (float*)alloc(256 * 4);
    float*    gsum   = (float*)alloc((size_t)NG * DD * 4);
    float*    gcnt   = (float*)alloc(NG * 4);
    _Float16* wf16   = (_Float16*)alloc((size_t)4 * 128 * 128 * 2);

    const float* hin = x;
    for (int L = 0; L < 3; ++L) {
        int ib = 3 + L * 10;
        const float* Wq = (const float*)d_in[ib + 0]; const float* bq = (const float*)d_in[ib + 1];
        const float* Wk = (const float*)d_in[ib + 2]; const float* bk = (const float*)d_in[ib + 3];
        const float* Wv = (const float*)d_in[ib + 4]; const float* bv = (const float*)d_in[ib + 5];
        const float* Ws = (const float*)d_in[ib + 6]; const float* bs = (const float*)d_in[ib + 7];
        const float* gamma = (const float*)d_in[ib + 8];
        const float* beta  = (const float*)d_in[ib + 9];

        k_pack_w<<<256, 256, 0, stream>>>(Wq, Wk, Wv, Ws, wf16);
        k_gemm<<<dim3(NN / 16, 4), 128, 0, stream>>>(hin, wf16, bq, bk, bv, bs,
                                                     Q, Km, V, OUT);
        k_fill_int<<<(NN + 255) / 256, 256, 0, stream>>>(maxb, 0x807fffff, NN); // enc(-inf)
        k_zero<<<(NN + 255) / 256, 256, 0, stream>>>(denom, NN);
        k_logits<<<EE / 8, 256, 0, stream>>>(Q, Km, src, dst, logits, maxb);
        k_exp<<<(EE + 255) / 256, 256, 0, stream>>>(logits, maxb, dst, wexp, denom);
        k_scatter<<<EE / 8, 256, 0, stream>>>(wexp, denom, V, src, dst, OUT);
        k_zero<<<1, 256, 0, stream>>>(bnsum, 256);
        k_bn_stats<<<(NN + 63) / 64, 256, 0, stream>>>(OUT, bnsum);
        k_bn_coef<<<1, 128, 0, stream>>>(bnsum, gamma, beta, coef);
        k_bn_relu<<<(NN * DD) / 256, 256, 0, stream>>>(OUT, coef, H);
        hin = H;
    }

    // gsum and gcnt are contiguous (gsum size is 256B-aligned) -> zero both at once
    k_zero<<<(NG * DD + NG + 255) / 256, 256, 0, stream>>>(gsum, NG * DD + NG);
    k_pool<<<(NN * DD) / 256, 256, 0, stream>>>(H, batch, gsum, gcnt);
    k_head<<<NG, 32, 0, stream>>>(gsum, gcnt, Wlin, blin, (float*)d_out);
}
